// GCGAT_v4pro_33732673143522
// MI455X (gfx1250) — matrix operand order repaired and checked by reading, not compiled, not run on hardware
//
#include <hip/hip_runtime.h>
#include <hip/hip_bf16.h>
#include <stdint.h>
#include <stddef.h>

// ---------------------------------------------------------------------------
// GCGAT (AttentiveFP x3 channels + MLP head) for MI455X / gfx1250.
// GEMMs: bf16 WMMA (v_wmma_f32_16x16x32_bf16) with f32 accumulate.
// Operands are pre-staged in bf16 (X row-major K-padded, W transposed to
// [n][k]) so WMMA fragments are direct contiguous b128 vector loads per the
// CDNA5 16-bit A/B VGPR layouts -- no LDS, no per-element bounds checks.
// Graph ops: float atomics + order-preserving uint atomicMax for seg-softmax.
// ---------------------------------------------------------------------------

#define ACT_NONE     0
#define ACT_LRELU01  1   // leaky_relu slope 0.01 (jax default)
#define ACT_LRELU001 2   // leaky_relu slope 0.001
#define ACT_RELU     3

typedef __attribute__((ext_vector_type(16))) __bf16 v16bf;
typedef __attribute__((ext_vector_type(8)))  __bf16 v8bf;
typedef __attribute__((ext_vector_type(8)))  float  v8f;

__device__ __forceinline__ float aact(float x, int act) {
  if (act == ACT_LRELU01)  return x > 0.f ? x : 0.01f  * x;
  if (act == ACT_LRELU001) return x > 0.f ? x : 0.001f * x;
  if (act == ACT_RELU)     return x > 0.f ? x : 0.f;
  return x;
}

__device__ __forceinline__ __bf16 f2bf(float f) {
  unsigned u = __float_as_uint(f);
  unsigned r = u + 0x7FFFu + ((u >> 16) & 1u);   // round-to-nearest-even
  unsigned short hs = (unsigned short)(r >> 16);
  __bf16 b;
  __builtin_memcpy(&b, &hs, 2);
  return b;
}

// order-preserving float<->uint for atomic max
__device__ __forceinline__ unsigned fenc(float f) {
  unsigned u = __float_as_uint(f);
  return (u >> 31) ? ~u : (u | 0x80000000u);
}
__device__ __forceinline__ float fdec(unsigned k) {
  unsigned u = (k & 0x80000000u) ? (k & 0x7FFFFFFFu) : ~k;
  return __uint_as_float(u);
}

// ---------------------------------------------------------------------------
// GEMM staging: X[M,K] f32 -> Xb[M,Kpad] bf16 (zero-padded K)
//               W[K,N] f32 -> Wt[Npad,Kpad] bf16 (transposed, zero-padded)
// ---------------------------------------------------------------------------
__global__ void k_prep_x(const float* __restrict__ X, __bf16* __restrict__ Xb,
                         size_t M, int K, int Kpad) {
  size_t i = (size_t)blockIdx.x * blockDim.x + threadIdx.x;
  if (i >= M * (size_t)Kpad) return;
  size_t m = i / (size_t)Kpad;
  int k = (int)(i % (size_t)Kpad);
  Xb[i] = (k < K) ? f2bf(X[m * K + k]) : f2bf(0.f);
}

__global__ void k_prep_w(const float* __restrict__ W, __bf16* __restrict__ Wt,
                         int K, int N, int Kpad, int Npad) {
  size_t i = (size_t)blockIdx.x * blockDim.x + threadIdx.x;
  if (i >= (size_t)Npad * Kpad) return;
  int n = (int)(i / (size_t)Kpad);
  int k = (int)(i % (size_t)Kpad);
  Wt[i] = (k < K && n < N) ? f2bf(W[(size_t)k * N + n]) : f2bf(0.f);
}

// ---------------------------------------------------------------------------
// WMMA GEMM:  Y[M,N] = act( X @ W + bias ).  One wave per 16x32 output block
// (two 16x16 N-tiles sharing the A fragment).  M must be a multiple of 16
// (true for every call site).  Fragments are direct vector loads:
//   A lane frag = Xb[row][k0+8h .. +7] ++ Xb[row][k0+16+8h .. +7]
//   B lane frag = Wt[n][k0+8h .. +7] ++ Wt[n][k0+16+8h .. +7]
// ---------------------------------------------------------------------------
__global__ void k_gemm_bf(const __bf16* __restrict__ Xb, const __bf16* __restrict__ Wt,
                          const float* __restrict__ B, float* __restrict__ Y,
                          int M, int Kpad, int N, int act) {
  const int lane = threadIdx.x;
  const int row  = lane & 15;
  const int half = lane >> 4;
  const int tm = blockIdx.y;
  const int n0 = blockIdx.x * 32;
  const __bf16* ax  = Xb + (size_t)(tm * 16 + row) * Kpad;
  const __bf16* bp0 = Wt + (size_t)(n0 + row) * Kpad + 8 * half;
  const __bf16* bp1 = bp0 + (size_t)16 * Kpad;
  v8f acc0 = {}, acc1 = {};
#pragma unroll 2
  for (int k0 = 0; k0 < Kpad; k0 += 32) {
    if (k0 + 32 < Kpad) __builtin_prefetch(ax + k0 + 32, 0, 1);
    v8bf alo = *(const v8bf*)(ax + k0 + 8 * half);
    v8bf ahi = *(const v8bf*)(ax + k0 + 16 + 8 * half);
    v16bf av = __builtin_shufflevector(alo, ahi, 0, 1, 2, 3, 4, 5, 6, 7,
                                       8, 9, 10, 11, 12, 13, 14, 15);
    v16bf bv0 = __builtin_shufflevector(*(const v8bf*)(bp0 + k0), *(const v8bf*)(bp0 + k0 + 16),
                                        0, 1, 2, 3, 4, 5, 6, 7, 8, 9, 10, 11, 12, 13, 14, 15);
    v16bf bv1 = __builtin_shufflevector(*(const v8bf*)(bp1 + k0), *(const v8bf*)(bp1 + k0 + 16),
                                        0, 1, 2, 3, 4, 5, 6, 7, 8, 9, 10, 11, 12, 13, 14, 15);
    acc0 = __builtin_amdgcn_wmma_f32_16x16x32_bf16(false, av, false, bv0,
                                                   (short)0, acc0, false, false);
    acc1 = __builtin_amdgcn_wmma_f32_16x16x32_bf16(false, av, false, bv1,
                                                   (short)0, acc1, false, false);
  }
  int gn0 = n0 + row, gn1 = n0 + 16 + row;
  float b0 = (B && gn0 < N) ? B[gn0] : 0.f;
  float b1 = (B && gn1 < N) ? B[gn1] : 0.f;
#pragma unroll
  for (int r = 0; r < 8; ++r) {
    int gm = tm * 16 + r + 8 * half;  // always < M (M % 16 == 0)
    if (gn0 < N) Y[(size_t)gm * N + gn0] = aact(acc0[r] + b0, act);
    if (gn1 < N) Y[(size_t)gm * N + gn1] = aact(acc1[r] + b1, act);
  }
}

// ---------------------------------------------------------------------------
// BatchNorm (training mode): per-column mean / rsqrt(var+eps) then affine+act
// ---------------------------------------------------------------------------
__global__ void k_colstats(const float* __restrict__ X, int M, int N,
                           float* __restrict__ mean, float* __restrict__ rstd) {
  __shared__ float s1[256], s2[256];
  int c = blockIdx.x;
  float a = 0.f, b = 0.f;
  for (int r = threadIdx.x; r < M; r += blockDim.x) {
    float v = X[(size_t)r * N + c];
    a += v; b += v * v;
  }
  s1[threadIdx.x] = a; s2[threadIdx.x] = b;
  __syncthreads();
  for (int off = 128; off > 0; off >>= 1) {
    if ((int)threadIdx.x < off) {
      s1[threadIdx.x] += s1[threadIdx.x + off];
      s2[threadIdx.x] += s2[threadIdx.x + off];
    }
    __syncthreads();
  }
  if (threadIdx.x == 0) {
    float mu = s1[0] / (float)M;
    float var = s2[0] / (float)M - mu * mu;
    mean[c] = mu;
    rstd[c] = rsqrtf(var + 1e-5f);
  }
}

__global__ void k_bn(const float* __restrict__ X, float* __restrict__ Y,
                     const float* __restrict__ g, const float* __restrict__ b,
                     const float* __restrict__ mean, const float* __restrict__ rstd,
                     size_t total, int N, int act) {
  size_t i = (size_t)blockIdx.x * blockDim.x + threadIdx.x;
  if (i >= total) return;
  int c = (int)(i % (size_t)N);
  float y = g[c] * (X[i] - mean[c]) * rstd[c] + b[c];
  Y[i] = aact(y, act);
}

__global__ void k_fill(float* __restrict__ p, size_t n, float v) {
  size_t i = (size_t)blockIdx.x * blockDim.x + threadIdx.x;
  if (i < n) p[i] = v;
}

// ---------------------------------------------------------------------------
// Edge attention score: s = lrelu( w . [h[dst]; h[src]; e] + b0 )  (HID = 128)
// ---------------------------------------------------------------------------
__global__ void k_att_edge(const float* __restrict__ h, const float* __restrict__ e,
                           const int* __restrict__ ei, int E,
                           const float* __restrict__ w, const float* __restrict__ b0,
                           float* __restrict__ s) {
  int idx = blockIdx.x * blockDim.x + threadIdx.x;
  if (idx >= E) return;
  int src = ei[idx], dst = ei[E + idx];
  const float4* hd = (const float4*)(h + (size_t)dst * 128);
  const float4* hs = (const float4*)(h + (size_t)src * 128);
  const float4* ev = (const float4*)(e + (size_t)idx * 128);
  const float4* w4 = (const float4*)w;
  float acc = b0[0];
#pragma unroll 8
  for (int j = 0; j < 32; ++j) { float4 a = hd[j], q = w4[j];      acc += a.x*q.x + a.y*q.y + a.z*q.z + a.w*q.w; }
#pragma unroll 8
  for (int j = 0; j < 32; ++j) { float4 a = hs[j], q = w4[32 + j]; acc += a.x*q.x + a.y*q.y + a.z*q.z + a.w*q.w; }
#pragma unroll 8
  for (int j = 0; j < 32; ++j) { float4 a = ev[j], q = w4[64 + j]; acc += a.x*q.x + a.y*q.y + a.z*q.z + a.w*q.w; }
  s[idx] = acc > 0.f ? acc : 0.01f * acc;
}

// Mol attention score: s = lrelu( w . [g[seg[n]]; h[n]] + b0 )
__global__ void k_att_node(const float* __restrict__ g, const float* __restrict__ h,
                           const int* __restrict__ seg, int n,
                           const float* __restrict__ w, const float* __restrict__ b0,
                           float* __restrict__ s) {
  int idx = blockIdx.x * blockDim.x + threadIdx.x;
  if (idx >= n) return;
  int sg = seg[idx];
  const float4* gv = (const float4*)(g + (size_t)sg * 128);
  const float4* hv = (const float4*)(h + (size_t)idx * 128);
  const float4* w4 = (const float4*)w;
  float acc = b0[0];
#pragma unroll 8
  for (int j = 0; j < 32; ++j) { float4 a = gv[j], q = w4[j];      acc += a.x*q.x + a.y*q.y + a.z*q.z + a.w*q.w; }
#pragma unroll 8
  for (int j = 0; j < 32; ++j) { float4 a = hv[j], q = w4[32 + j]; acc += a.x*q.x + a.y*q.y + a.z*q.z + a.w*q.w; }
  s[idx] = acc > 0.f ? acc : 0.01f * acc;
}

// segment softmax pieces
__global__ void k_segmax(const float* __restrict__ s, const int* __restrict__ seg,
                         unsigned* __restrict__ key, int n) {
  int i = blockIdx.x * blockDim.x + threadIdx.x;
  if (i >= n) return;
  atomicMax(&key[seg[i]], fenc(s[i]));
}
__global__ void k_expsum(const float* __restrict__ s, const int* __restrict__ seg,
                         const unsigned* __restrict__ key, float* __restrict__ den,
                         float* __restrict__ ec, int n) {
  int i = blockIdx.x * blockDim.x + threadIdx.x;
  if (i >= n) return;
  int sg = seg[i];
  float v = __expf(s[i] - fdec(key[sg]));
  ec[i] = v;
  atomicAdd(&den[sg], v);
}

// scatter: out[dst] += a_e * mh[src]   (128 feats, 4 per thread)
__global__ void k_scatter_edge(const float* __restrict__ mh, const int* __restrict__ ei, int E,
                               const float* __restrict__ ec, const float* __restrict__ den,
                               float* __restrict__ out) {
  int t = blockIdx.x * blockDim.x + threadIdx.x;
  int e = t >> 5;
  if (e >= E) return;
  int c4 = t & 31;
  int src = ei[e], dst = ei[E + e];
  float a = ec[e] / (den[dst] + 1e-16f);
  float4 mv = ((const float4*)(mh + (size_t)src * 128))[c4];
  float* o = out + (size_t)dst * 128 + (size_t)c4 * 4;
  atomicAdd(o + 0, a * mv.x); atomicAdd(o + 1, a * mv.y);
  atomicAdd(o + 2, a * mv.z); atomicAdd(o + 3, a * mv.w);
}

// scatter: out[seg[n]] += a_n * mh[n]
__global__ void k_scatter_node(const float* __restrict__ mh, const int* __restrict__ seg, int n,
                               const float* __restrict__ ec, const float* __restrict__ den,
                               float* __restrict__ out) {
  int t = blockIdx.x * blockDim.x + threadIdx.x;
  int r = t >> 5;
  if (r >= n) return;
  int c4 = t & 31;
  int sg = seg[r];
  float a = ec[r] / (den[sg] + 1e-16f);
  float4 mv = ((const float4*)(mh + (size_t)r * 128))[c4];
  float* o = out + (size_t)sg * 128 + (size_t)c4 * 4;
  atomicAdd(o + 0, a * mv.x); atomicAdd(o + 1, a * mv.y);
  atomicAdd(o + 2, a * mv.z); atomicAdd(o + 3, a * mv.w);
}

// generic segment feature sum: out[seg[row], :] += x[row, :]
__global__ void k_segsum(const float* __restrict__ x, const int* __restrict__ seg,
                         float* __restrict__ out, size_t M, int Nf) {
  size_t t = (size_t)blockIdx.x * blockDim.x + threadIdx.x;
  if (t >= M * (size_t)Nf) return;
  size_t r = t / (size_t)Nf;
  int c = (int)(t % (size_t)Nf);
  atomicAdd(out + (size_t)seg[r] * Nf + c, x[r * Nf + c]);
}

// torch GRUCell: h = (1-z)*tanh(in + r*hn) + z*h   (gates packed [r|z|n], D=128)
__global__ void k_gru(const float* __restrict__ gi, const float* __restrict__ gh,
                      float* __restrict__ h, size_t M) {
  size_t t = (size_t)blockIdx.x * blockDim.x + threadIdx.x;
  if (t >= M * 128) return;
  size_t r = t >> 7;
  int c = (int)(t & 127);
  const float* a = gi + r * 384;
  const float* q = gh + r * 384;
  float rr = 1.f / (1.f + __expf(-(a[c] + q[c])));
  float zz = 1.f / (1.f + __expf(-(a[128 + c] + q[128 + c])));
  float nn = tanhf(a[256 + c] + rr * q[256 + c]);
  float hc = h[r * 128 + c];
  h[r * 128 + c] = (1.f - zz) * nn + zz * hc;
}

__global__ void k_concat2(const float* __restrict__ A, int Na,
                          const float* __restrict__ B, int Nb,
                          float* __restrict__ Y, size_t M) {
  int tot = Na + Nb;
  size_t i = (size_t)blockIdx.x * blockDim.x + threadIdx.x;
  if (i >= M * (size_t)tot) return;
  size_t r = i / tot; int c = (int)(i % tot);
  Y[i] = (c < Na) ? A[r * Na + c] : B[r * Nb + (c - Na)];
}

__global__ void k_concat3(const float* __restrict__ A, int Na,
                          const float* __restrict__ B, int Nb,
                          const float* __restrict__ C, int Nc,
                          float* __restrict__ Y, size_t M) {
  int tot = Na + Nb + Nc;
  size_t i = (size_t)blockIdx.x * blockDim.x + threadIdx.x;
  if (i >= M * (size_t)tot) return;
  size_t r = i / tot; int c = (int)(i % tot);
  float v;
  if (c < Na) v = A[r * Na + c];
  else if (c < Na + Nb) v = B[r * Nb + (c - Na)];
  else v = C[r * Nc + (c - Na - Nb)];
  Y[i] = v;
}

// ---------------------------------------------------------------------------
// Host-side orchestration
// ---------------------------------------------------------------------------
namespace {

constexpr int N_ = 32768, E_ = 65536, NF_ = 65536, EF_ = 131072;
constexpr int F_ = 8192, EJ_ = 16384, G_ = 1024;

struct Lin { const float* w; const float* b; };
struct Gru { const float* wi; const float* wh; const float* bi; const float* bh; };
struct AfpLayer { Lin att; Gru gru; Lin msg; };
struct Afp { AfpLayer atom[3]; AfpLayer mol[3]; Lin edge, node, out; };

// JAX pytree (sorted dict keys) leaf order within one AFP head (54 leaves):
// atom[i]: att.b, att.w, gru.bh, gru.bi, gru.wh, gru.wi, msg.b, msg.w  (x3)
// edge.b, edge.w ; mol[i] (x3 like atom) ; node.b, node.w ; out.b, out.w
Afp decode_afp(void* const* d, int base) {
  auto P = [&](int i) { return (const float*)d[base + i]; };
  Afp a;
  for (int i = 0; i < 3; ++i) {
    int o = 8 * i;
    a.atom[i].att.b = P(o + 0); a.atom[i].att.w = P(o + 1);
    a.atom[i].gru.bh = P(o + 2); a.atom[i].gru.bi = P(o + 3);
    a.atom[i].gru.wh = P(o + 4); a.atom[i].gru.wi = P(o + 5);
    a.atom[i].msg.b = P(o + 6); a.atom[i].msg.w = P(o + 7);
  }
  a.edge.b = P(24); a.edge.w = P(25);
  for (int i = 0; i < 3; ++i) {
    int o = 26 + 8 * i;
    a.mol[i].att.b = P(o + 0); a.mol[i].att.w = P(o + 1);
    a.mol[i].gru.bh = P(o + 2); a.mol[i].gru.bi = P(o + 3);
    a.mol[i].gru.wh = P(o + 4); a.mol[i].gru.wi = P(o + 5);
    a.mol[i].msg.b = P(o + 6); a.mol[i].msg.w = P(o + 7);
  }
  a.node.b = P(50); a.node.w = P(51);
  a.out.b = P(52); a.out.w = P(53);
  return a;
}

struct Pool {
  float *h, *e, *mh, *macc, *gi, *gh, *s, *ec, *den, *g;
  unsigned* key;
  float* stats;   // mean @ +0, rstd @ +512
  __bf16* xb;     // bf16 staged activations [M, Kpad]
  __bf16* wt;     // bf16 transposed weights [Npad, Kpad]
};

inline unsigned blks(size_t n) { return (unsigned)((n + 255) / 256); }

// full GEMM: stage X->bf16, W->bf16 transposed, then WMMA kernel
void gemm(const float* X, const Lin& L, float* Y, int M, int K, int N, int act,
          const Pool& pl, hipStream_t st) {
  int Kpad = (K + 31) & ~31;
  int Npad = (N + 31) & ~31;
  k_prep_x<<<blks((size_t)M * Kpad), 256, 0, st>>>(X, pl.xb, (size_t)M, K, Kpad);
  k_prep_w<<<blks((size_t)Npad * Kpad), 256, 0, st>>>(L.w, pl.wt, K, N, Kpad, Npad);
  k_gemm_bf<<<dim3(Npad / 32, M / 16), dim3(32), 0, st>>>(pl.xb, pl.wt, L.b, Y,
                                                          M, Kpad, N, act);
}

void bn_fwd(const float* xin, float* yout, const float* g, const float* b,
            int M, int N, int act, float* stats, hipStream_t st) {
  k_colstats<<<N, 256, 0, st>>>(xin, M, N, stats, stats + 512);
  size_t tot = (size_t)M * N;
  k_bn<<<blks(tot), 256, 0, st>>>(xin, yout, g, b, stats, stats + 512, tot, N, act);
}

void run_afp(const Afp& P, const float* x, int nin, const int* ei,
             const float* ea, int ein, const int* seg, int n, int m, int ngr,
             int outdim, float* out, int out_act, const Pool& pl, hipStream_t st) {
  gemm(x, P.node, pl.h, n, nin, 128, ACT_LRELU01, pl, st);
  gemm(ea, P.edge, pl.e, m, ein, 128, ACT_LRELU01, pl, st);
  // --- atom message-passing layers ---
  for (int l = 0; l < 3; ++l) {
    const AfpLayer& L = P.atom[l];
    k_att_edge<<<blks(m), 256, 0, st>>>(pl.h, pl.e, ei, m, L.att.w, L.att.b, pl.s);
    k_fill<<<blks(n), 256, 0, st>>>((float*)pl.key, n, 0.f);
    k_segmax<<<blks(m), 256, 0, st>>>(pl.s, ei + m, pl.key, m);
    k_fill<<<blks(n), 256, 0, st>>>(pl.den, n, 0.f);
    k_expsum<<<blks(m), 256, 0, st>>>(pl.s, ei + m, pl.key, pl.den, pl.ec, m);
    gemm(pl.h, L.msg, pl.mh, n, 128, 128, ACT_NONE, pl, st);
    k_fill<<<blks((size_t)n * 128), 256, 0, st>>>(pl.macc, (size_t)n * 128, 0.f);
    k_scatter_edge<<<blks((size_t)m * 32), 256, 0, st>>>(pl.mh, ei, m, pl.ec, pl.den, pl.macc);
    gemm(pl.macc, Lin{L.gru.wi, L.gru.bi}, pl.gi, n, 128, 384, ACT_NONE, pl, st);
    gemm(pl.h,    Lin{L.gru.wh, L.gru.bh}, pl.gh, n, 128, 384, ACT_NONE, pl, st);
    k_gru<<<blks((size_t)n * 128), 256, 0, st>>>(pl.gi, pl.gh, pl.h, (size_t)n);
  }
  // --- super-node init + mol layers ---
  k_fill<<<blks((size_t)ngr * 128), 256, 0, st>>>(pl.g, (size_t)ngr * 128, 0.f);
  k_segsum<<<blks((size_t)n * 128), 256, 0, st>>>(pl.h, seg, pl.g, (size_t)n, 128);
  for (int l = 0; l < 3; ++l) {
    const AfpLayer& L = P.mol[l];
    k_att_node<<<blks(n), 256, 0, st>>>(pl.g, pl.h, seg, n, L.att.w, L.att.b, pl.s);
    k_fill<<<blks(ngr), 256, 0, st>>>((float*)pl.key, ngr, 0.f);
    k_segmax<<<blks(n), 256, 0, st>>>(pl.s, seg, pl.key, n);
    k_fill<<<blks(ngr), 256, 0, st>>>(pl.den, ngr, 0.f);
    k_expsum<<<blks(n), 256, 0, st>>>(pl.s, seg, pl.key, pl.den, pl.ec, n);
    gemm(pl.h, L.msg, pl.mh, n, 128, 128, ACT_NONE, pl, st);
    k_fill<<<blks((size_t)ngr * 128), 256, 0, st>>>(pl.macc, (size_t)ngr * 128, 0.f);
    k_scatter_node<<<blks((size_t)n * 32), 256, 0, st>>>(pl.mh, seg, n, pl.ec, pl.den, pl.macc);
    gemm(pl.macc, Lin{L.gru.wi, L.gru.bi}, pl.gi, ngr, 128, 384, ACT_NONE, pl, st);
    gemm(pl.g,    Lin{L.gru.wh, L.gru.bh}, pl.gh, ngr, 128, 384, ACT_NONE, pl, st);
    k_gru<<<blks((size_t)ngr * 128), 256, 0, st>>>(pl.gi, pl.gh, pl.g, (size_t)ngr);
  }
  gemm(pl.g, P.out, out, ngr, 128, outdim, out_act, pl, st);
}

}  // namespace

extern "C" void kernel_launch(void* const* d_in, const int* in_sizes, int n_in,
                              void* d_out, int out_size, void* d_ws, size_t ws_size,
                              hipStream_t stream) {
  (void)in_sizes; (void)n_in; (void)out_size;
  // --- data inputs ---
  const float* x      = (const float*)d_in[0];
  const int*   ei     = (const int*)  d_in[1];
  const float* eattr  = (const float*)d_in[2];
  const int*   batch  = (const int*)  d_in[3];
  const float* fx     = (const float*)d_in[4];
  const int*   fei    = (const int*)  d_in[5];
  const float* fea    = (const float*)d_in[6];
  const int*   fbatch = (const int*)  d_in[7];
  const float* jx     = (const float*)d_in[8];
  const int*   jei    = (const int*)  d_in[9];
  const float* jea    = (const float*)d_in[10];
  const int*   jbatch = (const int*)  d_in[11];
  auto P = [&](int i) { return (const float*)d_in[i]; };
  // param leaf bases (sorted-key pytree): frag@12, jt@70, origin@134, pred@200
  const Afp afp_frag   = decode_afp(d_in, 16);
  const Afp afp_jt     = decode_afp(d_in, 78);
  const Afp afp_origin = decode_afp(d_in, 142);

  // --- workspace carve ---
  size_t off = 0;
  auto alloc = [&](size_t bytes) -> float* {
    float* p = (float*)((char*)d_ws + off);
    off += (bytes + 511) & ~(size_t)511;
    return p;
  };
  Pool pl;
  pl.h    = alloc((size_t)NF_ * 128 * 4);
  pl.e    = alloc((size_t)EF_ * 128 * 4);
  pl.mh   = alloc((size_t)NF_ * 128 * 4);
  pl.macc = alloc((size_t)NF_ * 128 * 4);
  pl.gi   = alloc((size_t)NF_ * 384 * 4);
  pl.gh   = alloc((size_t)NF_ * 384 * 4);
  pl.s    = alloc((size_t)EF_ * 4);
  pl.ec   = alloc((size_t)EF_ * 4);
  pl.key  = (unsigned*)alloc((size_t)NF_ * 4);
  pl.den  = alloc((size_t)NF_ * 4);
  pl.g    = alloc((size_t)F_ * 128 * 4);
  pl.stats = alloc(1024 * 4);
  pl.xb   = (__bf16*)alloc((size_t)NF_ * 128 * 2);  // 16 MB bf16 staging (max M*Kpad)
  pl.wt   = (__bf16*)alloc(512 * 512 * 2);          // bf16 transposed weights (max 416x224)
  float* aout = alloc((size_t)F_ * 133 * 4);  // AFP head outputs (ho / hf / super_g)
  float* tmp  = alloc((size_t)F_ * 261 * 4);  // pre-BN temporaries, xj
  float* graph_origin = alloc((size_t)G_ * 128 * 4);
  float* graph_frag   = alloc((size_t)F_ * 133 * 4);
  float* mn   = alloc((size_t)F_ * 128 * 4);
  float* xp   = alloc((size_t)F_ * 128 * 4);
  float* fres = alloc((size_t)G_ * 133 * 4);
  float* cf   = alloc((size_t)G_ * 389 * 4);
  float* p1   = alloc((size_t)G_ * 194 * 4);
  float* p2   = alloc((size_t)G_ * 194 * 4);
  if (off > ws_size) return;

  // ===================== Origin channel =====================
  float* exbuf = pl.mh;  // free until first msg-GEMM inside AFP
  gemm(x, Lin{P(199), P(198)}, exbuf, N_, 44, 128, ACT_NONE, pl, stream);
  bn_fwd(exbuf, exbuf, P(197), P(196), N_, 128, ACT_LRELU01, pl.stats, stream);
  float* eebuf = pl.gi;  // free until first GRU-GEMM inside AFP
  gemm(eattr, Lin{P(141), P(140)}, eebuf, E_, 12, 128, ACT_NONE, pl, stream);
  bn_fwd(eebuf, eebuf, P(139), P(138), E_, 128, ACT_LRELU01, pl.stats, stream);
  run_afp(afp_origin, exbuf, 128, ei, eebuf, 128, batch, N_, E_, G_, 128, aout,
          ACT_NONE, pl, stream);
  gemm(aout, Lin{P(137), P(136)}, tmp, G_, 128, 128, ACT_NONE, pl, stream);
  bn_fwd(tmp, graph_origin, P(135), P(134), G_, 128, ACT_RELU, pl.stats, stream);

  // ===================== Fragment channel =====================
  run_afp(afp_frag, fx, 44, fei, fea, 12, fbatch, NF_, EF_, F_, 133, aout,
          ACT_NONE, pl, stream);
  gemm(aout, Lin{P(15), P(14)}, tmp, F_, 133, 133, ACT_NONE, pl, stream);
  bn_fwd(tmp, graph_frag, P(13), P(12), F_, 133, ACT_RELU, pl.stats, stream);

  // ===================== Junction-tree channel =====================
  gemm(jx, Lin{P(77), P(76)}, tmp, F_, 219, 128, ACT_NONE, pl, stream);
  bn_fwd(tmp, mn, P(75), P(74), F_, 128, ACT_LRELU01, pl.stats, stream);
  float* jebuf = pl.gi;
  gemm(jea, Lin{P(73), P(72)}, jebuf, EJ_, 12, 128, ACT_NONE, pl, stream);
  bn_fwd(jebuf, jebuf, P(71), P(70), EJ_, 128, ACT_LRELU01, pl.stats, stream);
  k_concat2<<<blks((size_t)F_ * 261), 256, 0, stream>>>(graph_frag, 133, mn, 128, tmp, (size_t)F_);
  gemm(tmp, Lin{P(133), P(132)}, xp, F_, 261, 128, ACT_NONE, pl, stream);
  // single head: super_g = relu(mean(f)) == relu(f)  -> fold relu into out GEMM
  run_afp(afp_jt, xp, 128, jei, jebuf, 128, jbatch, F_, EJ_, G_, 128, aout,
          ACT_RELU, pl, stream);

  // fragment residual scatter-add
  k_fill<<<blks((size_t)G_ * 133), 256, 0, stream>>>(fres, (size_t)G_ * 133, 0.f);
  k_segsum<<<blks((size_t)F_ * 133), 256, 0, stream>>>(graph_frag, jbatch, fres, (size_t)F_, 133);

  // ===================== Prediction head =====================
  k_concat3<<<blks((size_t)G_ * 389), 256, 0, stream>>>(graph_origin, 128, fres, 133,
                                                        aout, 128, cf, (size_t)G_);
  gemm(cf, Lin{P(203), P(202)}, p1, G_, 389, 194, ACT_LRELU001, pl, stream);
  bn_fwd(p1, p1, P(201), P(200), G_, 194, ACT_NONE, pl.stats, stream);
  gemm(p1, Lin{P(205), P(204)}, p2, G_, 194, 194, ACT_LRELU001, pl, stream);
  gemm(p2, Lin{P(207), P(206)}, (float*)d_out, G_, 194, 1, ACT_NONE, pl, stream);
}
